// ReverseLSTMLayer_66795331388078
// MI455X (gfx1250) — matrix operand order repaired and checked by reading, not compiled, not run on hardware
//
#include <hip/hip_runtime.h>
#include <hip/hip_bf16.h>

// ---------------------------------------------------------------------------
// ReverseLSTMLayer for MI455X (gfx1250, wave32, WMMA bf16 16x16x32)
//   Phase 0: convert W_ih / W_hh to bf16, init h double-buffer + barrier ctr
//   Phase 1: x_gates[T*B,4H] = input @ W_ih^T + (b_ih+b_hh)
//            (WMMA GEMM, double-buffered LDS software pipeline)
//   Phase 2: persistent kernel, 1024 sequential cell steps with device-wide
//            barrier; W_hh slice pinned in LDS (bf16), c kept in registers,
//            x_gates prefetched ahead of the recurrent GEMM each step.
// ---------------------------------------------------------------------------

#define T_STEPS 1024
#define BATCH   32
#define ISZ     1024
#define HSZ     1024
#define G4H     4096            // 4*HSZ
#define MTOT    (T_STEPS*BATCH) // 32768

typedef __attribute__((ext_vector_type(8)))  float   v8f;
typedef __attribute__((ext_vector_type(16))) __bf16  v16bf;
typedef __attribute__((ext_vector_type(8)))  __bf16  v8bf;

__device__ __forceinline__ float sigmoid_f(float x) {
    return 1.0f / (1.0f + __expf(-x));
}
__device__ __forceinline__ float tanh_f(float x) {
    x = fminf(fmaxf(x, -15.0f), 15.0f);
    float e = __expf(2.0f * x);
    return (e - 1.0f) / (e + 1.0f);
}

__device__ __forceinline__ void grid_barrier(unsigned int* cnt, unsigned int target) {
    __syncthreads();
    if (threadIdx.x == 0) {
        __threadfence();
        __hip_atomic_fetch_add(cnt, 1u, __ATOMIC_RELEASE, __HIP_MEMORY_SCOPE_AGENT);
        while (__hip_atomic_load(cnt, __ATOMIC_ACQUIRE, __HIP_MEMORY_SCOPE_AGENT) < target) {
            __builtin_amdgcn_s_sleep(1);
        }
    }
    __syncthreads();
}

// ---------------- Phase 0: weight conversion + state init -------------------

__global__ __launch_bounds__(256) void k_convert_weights(
        const float* __restrict__ wih_f, const float* __restrict__ whh_f,
        __bf16* __restrict__ wih_b, __bf16* __restrict__ whh_b) {
    size_t i = (size_t)blockIdx.x * 256 + threadIdx.x;   // 4M threads exactly
    wih_b[i] = (__bf16)wih_f[i];
    whh_b[i] = (__bf16)whh_f[i];
}

__global__ __launch_bounds__(256) void k_init_state(
        const float* __restrict__ h0, __bf16* __restrict__ h_buf,
        unsigned int* __restrict__ cnt) {
    int i = blockIdx.x * 256 + threadIdx.x;              // 32768 threads
    h_buf[i] = (__bf16)h0[i];
    if (i == 0) *cnt = 0u;
}

// ---------------- Phase 1: x_gates GEMM (M=32768, N=4096, K=1024) -----------
// Block tile 64(M) x 128(N), 8 waves in a 2x4 grid, each wave 32x32 output.
// Double-buffered LDS: global loads for chunk k+1 overlap WMMAs of chunk k.

__global__ __launch_bounds__(256) void k_xgates(
        const float* __restrict__ input,     // [MTOT, ISZ] fp32
        const __bf16* __restrict__ wih,      // [G4H, ISZ]  bf16
        const float* __restrict__ b_ih,
        const float* __restrict__ b_hh,
        float* __restrict__ xg) {            // [MTOT, G4H] fp32
    __shared__ __bf16 A_lds[2][64][32];      //  8 KB
    __shared__ __bf16 B_lds[2][128][32];     // 16 KB

    const int tid  = threadIdx.x;
    const int lane = tid & 31;
    const int w    = tid >> 5;
    const int wm   = w >> 2;          // 0..1
    const int wn   = w & 3;           // 0..3
    const int half = lane >> 4;
    const int l16  = lane & 15;
    const int Mbase = blockIdx.y * 64;
    const int Nbase = blockIdx.x * 128;

    // per-thread staging roles
    const int ar = tid >> 2, aq = tid & 3;        // A: 4 threads/row, 8 f32 each
    const int bn = tid >> 1, bh = tid & 1;        // B: 2 threads/row, 16 bf16 each
    const float* a_src  = input + (size_t)(Mbase + ar) * ISZ + aq * 8;
    const __bf16* b_src = wih + (size_t)(Nbase + bn) * ISZ + bh * 16;

    v8f acc[2][2] = {};

    // prologue: stage chunk 0 into buffer 0
    {
        float4 f0 = *(const float4*)(a_src);
        float4 f1 = *(const float4*)(a_src + 4);
        __bf16* dst = &A_lds[0][ar][aq * 8];
        dst[0] = (__bf16)f0.x; dst[1] = (__bf16)f0.y;
        dst[2] = (__bf16)f0.z; dst[3] = (__bf16)f0.w;
        dst[4] = (__bf16)f1.x; dst[5] = (__bf16)f1.y;
        dst[6] = (__bf16)f1.z; dst[7] = (__bf16)f1.w;
        *(v16bf*)&B_lds[0][bn][bh * 16] = *(const v16bf*)b_src;
    }
    __syncthreads();

    for (int kc = 0; kc < 32; ++kc) {
        const int cur = kc & 1, nxt = cur ^ 1;

        // issue global loads for chunk kc+1 (latency hidden under WMMAs)
        float4 f0, f1; v16bf bv;
        const bool more = (kc < 31);
        if (more) {
            const int k1 = (kc + 1) * 32;
            f0 = *(const float4*)(a_src + k1);
            f1 = *(const float4*)(a_src + k1 + 4);
            bv = *(const v16bf*)(b_src + k1);
        }

        // compute on buffer `cur`
        v16bf afr[2];
        #pragma unroll
        for (int tM = 0; tM < 2; ++tM) {
            const __bf16* ap = &A_lds[cur][wm * 32 + tM * 16 + l16][half * 16];
            v8bf alo = *(const v8bf*)ap;
            v8bf ahi = *(const v8bf*)(ap + 8);
            #pragma unroll
            for (int i = 0; i < 8; ++i) { afr[tM][i] = alo[i]; afr[tM][i + 8] = ahi[i]; }
        }
        v16bf bfr[2];
        #pragma unroll
        for (int tN = 0; tN < 2; ++tN)
            bfr[tN] = *(const v16bf*)&B_lds[cur][wn * 32 + tN * 16 + l16][half * 16];

        #pragma unroll
        for (int tM = 0; tM < 2; ++tM)
            #pragma unroll
            for (int tN = 0; tN < 2; ++tN)
                acc[tM][tN] = __builtin_amdgcn_wmma_f32_16x16x32_bf16(
                    false, afr[tM], false, bfr[tN], (short)0, acc[tM][tN],
                    false, false);

        // stage chunk kc+1 into buffer `nxt`
        if (more) {
            __bf16* dst = &A_lds[nxt][ar][aq * 8];
            dst[0] = (__bf16)f0.x; dst[1] = (__bf16)f0.y;
            dst[2] = (__bf16)f0.z; dst[3] = (__bf16)f0.w;
            dst[4] = (__bf16)f1.x; dst[5] = (__bf16)f1.y;
            dst[6] = (__bf16)f1.z; dst[7] = (__bf16)f1.w;
            *(v16bf*)&B_lds[nxt][bn][bh * 16] = bv;
        }
        __syncthreads();
    }

    // epilogue: add fused bias, store fp32
    #pragma unroll
    for (int tM = 0; tM < 2; ++tM) {
        #pragma unroll
        for (int tN = 0; tN < 2; ++tN) {
            const int col  = Nbase + wn * 32 + tN * 16 + l16;
            const float bias = b_ih[col] + b_hh[col];
            const int row0 = Mbase + wm * 32 + tM * 16 + half * 8;
            #pragma unroll
            for (int r = 0; r < 8; ++r)
                xg[(size_t)(row0 + r) * G4H + col] = acc[tM][tN][r] + bias;
        }
    }
}

// ---------------- Phase 2: persistent reverse LSTM scan ---------------------
// 64 blocks x 256 threads. Block b owns hidden units j = b*16 .. b*16+15 and
// ALL FOUR gate columns for them; W_hh slice (64 x 1024 bf16 = 128KB) lives in
// LDS for the whole scan. One device-wide barrier per step; h double-buffered;
// x_gates for the step prefetched into registers before the recurrent GEMM.

__global__ __launch_bounds__(256) void k_lstm_scan(
        const float* __restrict__ xg,        // [T,B,4H] fp32 (biases folded)
        const __bf16* __restrict__ whh,      // [4H, H] bf16
        const float* __restrict__ c0,        // [B, H]
        __bf16* __restrict__ h_buf,          // [2][B*H] bf16 ping-pong
        float* __restrict__ out,             // [T*B*H] ++ h_fin ++ c_fin
        unsigned int* __restrict__ cnt,
        int nblocks) {
    __shared__ __bf16 Wl[64][1024];          // 128 KB: gate-major rows g*16+j
    __shared__ __bf16 Hl[32][1024];          //  64 KB: current h (bf16)
    __shared__ float  Gl[4][32][16];         //   8 KB: recurrent gate partials

    const int tid  = threadIdx.x;
    const int lane = tid & 31;
    const int w    = tid >> 5;
    const int mt   = w & 1;                  // M tile (batch 0-15 / 16-31)
    const int nt   = w >> 1;                 // gate index 0..3
    const int half = lane >> 4;
    const int l16  = lane & 15;
    const int jbase = blockIdx.x * 16;

    // stage W_hh slice once (4 threads/row, 512B each)
    {
        const int r = tid >> 2, q = tid & 3;
        const int n = (r >> 4) * HSZ + jbase + (r & 15);   // global gate row
        const uint4* src = (const uint4*)(whh + (size_t)n * HSZ) + q * 32;
        uint4* dst = (uint4*)&Wl[r][q * 256];
        #pragma unroll 8
        for (int i = 0; i < 32; ++i) dst[i] = src[i];
    }

    // per-thread cell state: 2 elements (b, j) held in registers for all steps
    const int e0 = tid, e1 = tid + 256;
    const int b0 = e0 >> 4, j0 = e0 & 15;
    const int b1 = e1 >> 4, j1 = e1 & 15;
    float creg0 = c0[b0 * HSZ + jbase + j0];
    float creg1 = c0[b1 * HSZ + jbase + j1];

    for (int s = 0; s < T_STEPS; ++s) {
        const int t = T_STEPS - 1 - s;       // reverse scan
        const int p = s & 1;

        // prefetch this step's x_gates (independent of the h chain); the
        // s_wait for these lands after the 32-WMMA GEMM below.
        const size_t xi0 = (size_t)t * (BATCH * G4H) + (size_t)b0 * G4H + jbase + j0;
        const size_t xi1 = (size_t)t * (BATCH * G4H) + (size_t)b1 * G4H + jbase + j1;
        float xr0[4], xr1[4];
        #pragma unroll
        for (int g = 0; g < 4; ++g) {
            xr0[g] = xg[xi0 + (size_t)g * HSZ];
            xr1[g] = xg[xi1 + (size_t)g * HSZ];
        }

        // stage h (bf16, 64KB) from global double buffer
        {
            const uint4* src = (const uint4*)(h_buf + (size_t)p * (BATCH * HSZ));
            uint4* dst = (uint4*)&Hl[0][0];
            #pragma unroll
            for (int i = 0; i < 16; ++i) dst[tid + i * 256] = src[tid + i * 256];
        }
        __syncthreads();

        // recurrent GEMM: this wave's 16x16 tile, full K=1024
        v8f acc = {};
        #pragma unroll 4
        for (int kc = 0; kc < 32; ++kc) {
            const int k = kc * 32;
            const __bf16* ap = &Hl[mt * 16 + l16][k + half * 16];
            v8bf alo = *(const v8bf*)ap;
            v8bf ahi = *(const v8bf*)(ap + 8);
            v16bf afr;
            #pragma unroll
            for (int i = 0; i < 8; ++i) { afr[i] = alo[i]; afr[i + 8] = ahi[i]; }
            v16bf bfr = *(const v16bf*)&Wl[nt * 16 + l16][k + half * 16];
            acc = __builtin_amdgcn_wmma_f32_16x16x32_bf16(
                false, afr, false, bfr, (short)0, acc, false, false);
        }
        #pragma unroll
        for (int r = 0; r < 8; ++r)
            Gl[nt][mt * 16 + half * 8 + r][l16] = acc[r];
        __syncthreads();

        // elementwise cell update (2 elements per thread, c in registers)
        #pragma unroll
        for (int e = 0; e < 2; ++e) {
            const int b = e ? b1 : b0;
            const int j = e ? j1 : j0;
            const float* xr = e ? xr1 : xr0;
            float iv = Gl[0][b][j] + xr[0];
            float fv = Gl[1][b][j] + xr[1];
            float gv = Gl[2][b][j] + xr[2];
            float ov = Gl[3][b][j] + xr[3];
            iv = sigmoid_f(iv); fv = sigmoid_f(fv);
            gv = tanh_f(gv);    ov = sigmoid_f(ov);
            float c = fv * (e ? creg1 : creg0) + iv * gv;
            float hn = ov * tanh_f(c);
            if (e) creg1 = c; else creg0 = c;

            const size_t oi = (size_t)t * (BATCH * HSZ) + (size_t)b * HSZ + jbase + j;
            out[oi] = hn;
            h_buf[(size_t)(p ^ 1) * (BATCH * HSZ) + b * HSZ + jbase + j] = (__bf16)hn;
            if (t == 0) {
                const size_t fb = (size_t)T_STEPS * BATCH * HSZ;
                out[fb + b * HSZ + jbase + j] = hn;                   // h_fin
                out[fb + BATCH * HSZ + b * HSZ + jbase + j] = c;      // c_fin
            }
        }
        grid_barrier(cnt, (unsigned int)(s + 1) * (unsigned int)nblocks);
        __syncthreads();   // Hl/Gl safe to overwrite next iteration
    }
}

// ---------------------------------------------------------------------------

extern "C" void kernel_launch(void* const* d_in, const int* in_sizes, int n_in,
                              void* d_out, int out_size, void* d_ws, size_t ws_size,
                              hipStream_t stream) {
    const float* input = (const float*)d_in[0];   // [T,B,I]
    const float* h0    = (const float*)d_in[1];   // [B,H]
    const float* c0    = (const float*)d_in[2];   // [B,H]
    const float* W_ih  = (const float*)d_in[3];   // [4H,I]
    const float* W_hh  = (const float*)d_in[4];   // [4H,H]
    const float* b_ih  = (const float*)d_in[5];   // [4H]
    const float* b_hh  = (const float*)d_in[6];   // [4H]
    float* out = (float*)d_out;

    char* ws = (char*)d_ws;
    const size_t XG_BYTES = (size_t)MTOT * G4H * sizeof(float);   // 512 MiB
    const size_t W_BYTES  = (size_t)G4H * ISZ * sizeof(__bf16);   //   8 MiB
    float*        xg    = (float*)ws;
    __bf16*       wih_b = (__bf16*)(ws + XG_BYTES);
    __bf16*       whh_b = (__bf16*)(ws + XG_BYTES + W_BYTES);
    __bf16*       h_buf = (__bf16*)(ws + XG_BYTES + 2 * W_BYTES);   // 2*64KB
    unsigned int* cnt   = (unsigned int*)(ws + XG_BYTES + 2 * W_BYTES + 131072);

    // Phase 0
    k_convert_weights<<<(G4H * ISZ) / 256, 256, 0, stream>>>(W_ih, W_hh, wih_b, whh_b);
    k_init_state<<<(BATCH * HSZ) / 256, 256, 0, stream>>>(h0, h_buf, cnt);
    // Phase 1: x_gates GEMM
    k_xgates<<<dim3(G4H / 128, MTOT / 64), 256, 0, stream>>>(input, wih_b, b_ih, b_hh, xg);
    // Phase 2: persistent reverse scan (64 blocks == HSZ/16)
    k_lstm_scan<<<HSZ / 16, 256, 0, stream>>>(xg, whh_b, c0, h_buf, out, cnt, HSZ / 16);
}
